// KNNPNSDFDecoder_89756226552176
// MI455X (gfx1250) — matrix operand order repaired and checked by reading, not compiled, not run on hardware
//
#include <hip/hip_runtime.h>
#include <hip/hip_bf16.h>

typedef __attribute__((ext_vector_type(16))) _Float16 v16h;
typedef __attribute__((ext_vector_type(8)))  _Float16 v8h;
typedef __attribute__((ext_vector_type(8)))  float    v8f;

#define BB 2
#define NN 8192
#define MM 2048
#define CC 128
#define KK 16
#define ROWS (BB*NN)

// ---------------- workspace layout ----------------
// fp16 folded weights (half-element offsets, base = ws)
#define OFF_WPD1   0        // 128x32
#define OFF_WPD2   4096     // 128x128
#define OFF_WKNN1  20480    // 128x384
#define OFF_WKNN2  69632    // 128x128
#define OFF_WPT1   86016    // 128x128
#define OFF_WPT2   102400   // 128x128
#define OFF_WBLK1  118784   // 128x288
#define OFF_WAKNN  155648   // 128x128
#define OFF_WAPT   172032   // 128x128
#define WEIGHT_BYTES 376832
// fp32 folded biases (float offsets, base = ws + WEIGHT_BYTES)
#define BIA_PD1  0
#define BIA_PD2  128
#define BIA_KNN1 256
#define BIA_KNN2 384
#define BIA_PT1  512
#define BIA_PT2  640
#define BIA_BLK1 768
#define BIA_AKNN 896
#define BIA_APT  1024
// bulk intermediates (byte offsets)
#define OFFB_IDXK   381440                    // int   [ROWS][16]
#define OFFB_PDIFF  1430016                   // f16   [ROWS][16][4]
#define OFFB_VINT   3527168                   // f16   [ROWS][128]
#define OFFB_VP     7721472                   // f16   [ROWS][128]
#define OFFB_FPT    11915776                  // f16   [ROWS][128]

__device__ __forceinline__ float lrelu(float x) { return x >= 0.f ? x : 0.05f * x; }

__device__ __forceinline__ unsigned long long shflxor_u64(unsigned long long v, int m) {
    unsigned lo = (unsigned)v, hi = (unsigned)(v >> 32);
    lo = (unsigned)__shfl_xor((int)lo, m, 32);
    hi = (unsigned)__shfl_xor((int)hi, m, 32);
    return ((unsigned long long)hi << 32) | lo;
}

// ---------------------------------------------------------------------------
// Multi-slice WMMA GEMM: for each of NS column-slices,
//   act_out[s] = lrelu( W'(128 x cinPad) * act_in[s] + bias )
// One A-fragment load feeds NS v_wmma ops (halves weight traffic per slice).
// A-frag: lane = row M (mod 16), lane-half selects K 0..15 / 16..31 (contiguous).
// B-frag: lane = col, lane-half selects K 0..15 / 16..31 -> contiguous LDS.
// ---------------------------------------------------------------------------
template<int NS>
__device__ __forceinline__ void gemm_cbn(const _Float16* __restrict__ W,
                                         const float* __restrict__ bias,
                                         _Float16* act, int actStride, int sliceElems,
                                         int inOff, int cinPad, int outOff,
                                         bool relu, int lane)
{
    const int col = lane & 15;
    const int hi  = lane >> 4;
    for (int mt = 0; mt < 8; ++mt) {
        v8f acc[NS] = {};
        const _Float16* arow = W + (mt * 16 + col) * cinPad + hi * 16;
        for (int kt = 0; kt < cinPad; kt += 32) {
            v8h lo = *(const v8h*)(arow + kt);
            v8h hp = *(const v8h*)(arow + kt + 8);
            v16h a;
#pragma unroll
            for (int i = 0; i < 8; ++i) { a[i] = lo[i]; a[8 + i] = hp[i]; }
#pragma unroll
            for (int s = 0; s < NS; ++s) {
                v16h b = *(const v16h*)(act + s * sliceElems + col * actStride
                                        + inOff + hi * 16 + kt);
                acc[s] = __builtin_amdgcn_wmma_f32_16x16x32_f16(
                    false, a, false, b, (short)0, acc[s], false, false);
            }
        }
        const int rbase = mt * 16 + hi * 8;
#pragma unroll
        for (int s = 0; s < NS; ++s) {
#pragma unroll
            for (int r = 0; r < 8; ++r) {
                float v = acc[s][r] + bias[rbase + r];
                if (relu) v = lrelu(v);
                act[s * sliceElems + col * actStride + outOff + rbase + r] = (_Float16)v;
            }
        }
    }
}

// ---------------------------------------------------------------------------
// K0: fold BN into conv weights, convert to fp16, zero-pad Cin.
// ---------------------------------------------------------------------------
__global__ void fold_kernel(const float* __restrict__ w, const float* __restrict__ g,
                            const float* __restrict__ b, const float* __restrict__ m,
                            const float* __restrict__ v,
                            _Float16* __restrict__ wOut, float* __restrict__ bOut,
                            int cout, int cin, int cinPad)
{
    int i = blockIdx.x * blockDim.x + threadIdx.x;
    if (i >= cout * cinPad) return;
    int o = i / cinPad, c = i - o * cinPad;
    float s = g[o] * rsqrtf(v[o] + 1e-5f);
    wOut[o * cinPad + c] = (_Float16)(c < cin ? w[o * cin + c] * s : 0.f);
    if (c == 0) bOut[o] = b[o] - m[o] * s;
}

// ---------------------------------------------------------------------------
// K1: distances + top-16 selection + 3-NN interpolation.
// One 256-thread block per (b,n).  Distances live in registers (8/thread);
// argmin = packed (dist<<32|idx) u64 min: wave shfl butterfly + 8-way LDS merge.
// ---------------------------------------------------------------------------
__global__ __launch_bounds__(256) void knn_kernel(
    const float* __restrict__ p, const float* __restrict__ inputs,
    const float* __restrict__ feat_pt,
    int* __restrict__ idxK, _Float16* __restrict__ pdiffW, _Float16* __restrict__ vintW)
{
    __shared__ unsigned long long wred[8];
    __shared__ int   selI[KK];
    __shared__ float selD[KK];
    const int tid = threadIdx.x, lane = tid & 31, wv = tid >> 5;
    const int row = blockIdx.x;
    const int b = row >> 13;
    const float px = p[row * 3 + 0], py = p[row * 3 + 1], pz = p[row * 3 + 2];
    const float* inb = inputs + (size_t)b * MM * 3;
    float d[8];
#pragma unroll
    for (int i = 0; i < 8; ++i) {
        int j = tid + (i << 8);
        float dx = px - inb[j * 3 + 0];
        float dy = py - inb[j * 3 + 1];
        float dz = pz - inb[j * 3 + 2];
        d[i] = dx * dx + dy * dy + dz * dz;
    }
    for (int s = 0; s < KK; ++s) {
        unsigned long long best = ~0ull;
#pragma unroll
        for (int i = 0; i < 8; ++i) {
            int j = tid + (i << 8);
            unsigned long long pv =
                ((unsigned long long)__float_as_uint(d[i]) << 32) | (unsigned)j;
            best = pv < best ? pv : best;
        }
#pragma unroll
        for (int off = 16; off; off >>= 1) {
            unsigned long long o = shflxor_u64(best, off);
            best = o < best ? o : best;
        }
        if (lane == 0) wred[wv] = best;
        __syncthreads();
        unsigned long long g = wred[0];
#pragma unroll
        for (int w = 1; w < 8; ++w) { unsigned long long o = wred[w]; g = o < g ? o : g; }
        int idx = (int)(unsigned)g;
        if ((idx & 255) == tid) d[idx >> 8] = 3.4e38f;
        if (tid == 0) { selI[s] = idx; selD[s] = __uint_as_float((unsigned)(g >> 32)); }
        __syncthreads();
    }
    if (tid < KK) {
        int idx = selI[tid];
        idxK[row * KK + tid] = idx;
        _Float16* pd = pdiffW + (size_t)(row * KK + tid) * 4;
        pd[0] = (_Float16)(px - inb[idx * 3 + 0]);
        pd[1] = (_Float16)(py - inb[idx * 3 + 1]);
        pd[2] = (_Float16)(pz - inb[idx * 3 + 2]);
        pd[3] = (_Float16)0.f;
    }
    if (tid < CC) {
        float r0 = 1.f / (selD[0] + 1e-8f);
        float r1 = 1.f / (selD[1] + 1e-8f);
        float r2 = 1.f / (selD[2] + 1e-8f);
        float inv = 1.f / (r0 + r1 + r2);
        const float* fb = feat_pt + (size_t)b * MM * CC;
        float acc = r0 * inv * fb[(size_t)selI[0] * CC + tid]
                  + r1 * inv * fb[(size_t)selI[1] * CC + tid]
                  + r2 * inv * fb[(size_t)selI[2] * CC + tid];
        vintW[(size_t)row * CC + tid] = (_Float16)acc;
    }
}

// ---------------------------------------------------------------------------
// K2: per-n chain (pt1 -> pt2 -> vp ; attn_pt -> f_pt).
// One wave per block, 2 slices x 16 n-columns = 32 n's per block.
// ---------------------------------------------------------------------------
#define K2_STRIDE 256
#define K2_SLICE  (16 * K2_STRIDE)
__global__ __launch_bounds__(32) void pt_kernel(
    const _Float16* __restrict__ wsW, const float* __restrict__ wsB,
    const _Float16* __restrict__ vint,
    _Float16* __restrict__ vpW, _Float16* __restrict__ fptW)
{
    __shared__ __align__(32) _Float16 act[2][16][K2_STRIDE];
    const int lane = threadIdx.x & 31;
    const int col = lane & 15, hi = lane >> 4;
    _Float16* A = &act[0][0][0];
#pragma unroll
    for (int s = 0; s < 2; ++s) {
        int row = blockIdx.x * 32 + s * 16 + col;
        const _Float16* src = vint + (size_t)row * CC + hi * 64;
        for (int i = 0; i < 64; ++i)
            A[s * K2_SLICE + col * K2_STRIDE + hi * 64 + i] = src[i];
    }
    __syncthreads();
    gemm_cbn<2>(wsW + OFF_WPT1, wsB + BIA_PT1, A, K2_STRIDE, K2_SLICE, 0,   128, 128, true, lane);
    __syncthreads();
    gemm_cbn<2>(wsW + OFF_WPT2, wsB + BIA_PT2, A, K2_STRIDE, K2_SLICE, 128, 128, 0,   true, lane);
    __syncthreads();
#pragma unroll
    for (int s = 0; s < 2; ++s) {
        int row = blockIdx.x * 32 + s * 16 + col;
        _Float16* dst = vpW + (size_t)row * CC + hi * 64;
        for (int i = 0; i < 64; ++i)
            dst[i] = A[s * K2_SLICE + col * K2_STRIDE + hi * 64 + i];
    }
    gemm_cbn<2>(wsW + OFF_WAPT, wsB + BIA_APT, A, K2_STRIDE, K2_SLICE, 0, 128, 128, true, lane);
    __syncthreads();
#pragma unroll
    for (int s = 0; s < 2; ++s) {
        int row = blockIdx.x * 32 + s * 16 + col;
        _Float16* dst = fptW + (size_t)row * CC + hi * 64;
        for (int i = 0; i < 64; ++i)
            dst[i] = A[s * K2_SLICE + col * K2_STRIDE + 128 + hi * 64 + i];
    }
}

// ---------------------------------------------------------------------------
// K3: fused per-point chain.  One wave per block handles 2 points (slices);
// each point's 16 neighbors are the 16 WMMA columns.
// Per-slice activation buffer [16][512]:
//   [0:128) pfeat / vk, [128:256) vfeat_pdiff / vp / f_knn,
//   [256:288) pdiff(+pad), [256:384) fg, [384:512) GEMM scratch.
// ---------------------------------------------------------------------------
#define K3_STRIDE 512
#define K3_SLICE  (16 * K3_STRIDE)
__global__ __launch_bounds__(32) void fuse_kernel(
    const float* __restrict__ feat_g, const float* __restrict__ feat_pt,
    const float* __restrict__ w2, const float* __restrict__ b2,
    const _Float16* __restrict__ wsW, const float* __restrict__ wsB,
    const int* __restrict__ idxK, const _Float16* __restrict__ pdiffW,
    const _Float16* __restrict__ vpW, const _Float16* __restrict__ fptW,
    float* __restrict__ out)
{
    __shared__ __align__(32) _Float16 act[2][16][K3_STRIDE];
    const int lane = threadIdx.x & 31;
    const int r0 = blockIdx.x * 2;
    const int b = r0 >> 13;                  // pair never crosses batch boundary
    const int col = lane & 15, hi = lane >> 4;
    _Float16* A = &act[0][0][0];

    // warm L2/WGP$ for the big weight blocks (global_prefetch_b8)
    for (int off = lane * 64; off < 49152; off += 32 * 64)
        __builtin_prefetch((const void*)(wsW + OFF_WKNN1 + off), 0, 1);
    for (int off = lane * 64; off < 36864; off += 32 * 64)
        __builtin_prefetch((const void*)(wsW + OFF_WBLK1 + off), 0, 1);

    // phase 0: pdiff (padded to 32) at [256:288)
#pragma unroll
    for (int s = 0; s < 2; ++s) {
        const _Float16* pd = pdiffW + (size_t)((r0 + s) * KK + col) * 4;
        if (hi == 0) {
            _Float16* dst = A + s * K3_SLICE + col * K3_STRIDE;
            dst[256] = pd[0]; dst[257] = pd[1]; dst[258] = pd[2];
            for (int i = 259; i < 288; ++i) dst[i] = (_Float16)0.f;
        }
    }
    __syncthreads();
    gemm_cbn<2>(wsW + OFF_WPD1, wsB + BIA_PD1, A, K3_STRIDE, K3_SLICE, 256, 32,  384, true, lane);
    __syncthreads();
    gemm_cbn<2>(wsW + OFF_WPD2, wsB + BIA_PD2, A, K3_STRIDE, K3_SLICE, 384, 128, 128, true, lane);
    __syncthreads();
    // phase 3: gather pfeat -> [0:128), broadcast feat_g -> [256:384)
#pragma unroll
    for (int s = 0; s < 2; ++s) {
        int idx = idxK[(r0 + s) * KK + col];
        const float* src = feat_pt + ((size_t)b * MM + idx) * CC + hi * 64;
        _Float16* dst = A + s * K3_SLICE + col * K3_STRIDE;
        for (int i = 0; i < 64; ++i) dst[hi * 64 + i] = (_Float16)src[i];
        const float* fg = feat_g + b * CC + hi * 64;
        for (int i = 0; i < 64; ++i) dst[256 + hi * 64 + i] = (_Float16)fg[i];
    }
    __syncthreads();
    gemm_cbn<2>(wsW + OFF_WKNN1, wsB + BIA_KNN1, A, K3_STRIDE, K3_SLICE, 0,   384, 384, true, lane);
    __syncthreads();
    gemm_cbn<2>(wsW + OFF_WKNN2, wsB + BIA_KNN2, A, K3_STRIDE, K3_SLICE, 384, 128, 0,   true, lane);
    __syncthreads();
    // phase 6: vp -> [128:256), pdiff+pad -> [256:288)
#pragma unroll
    for (int s = 0; s < 2; ++s) {
        const _Float16* vp = vpW + (size_t)(r0 + s) * CC + hi * 64;
        _Float16* dst = A + s * K3_SLICE + col * K3_STRIDE;
        for (int i = 0; i < 64; ++i) dst[128 + hi * 64 + i] = vp[i];
        const _Float16* pd = pdiffW + (size_t)((r0 + s) * KK + col) * 4;
        if (hi == 0) {
            dst[256] = pd[0]; dst[257] = pd[1]; dst[258] = pd[2];
            for (int i = 259; i < 288; ++i) dst[i] = (_Float16)0.f;
        }
    }
    __syncthreads();
    gemm_cbn<2>(wsW + OFF_WBLK1, wsB + BIA_BLK1, A, K3_STRIDE, K3_SLICE, 0, 288, 384, true, lane);
    __syncthreads();
    // blk2: net_interp = dot(w2, h) + b2  (halves combined across lane^16)
    float net[2];
#pragma unroll
    for (int s = 0; s < 2; ++s) {
        float v = 0.f;
        const float* wv = w2 + hi * 64;
        const _Float16* h = A + s * K3_SLICE + col * K3_STRIDE + 384 + hi * 64;
        for (int i = 0; i < 64; ++i) v += (float)h[i] * wv[i];
        v += __shfl_xor(v, 16, 32);
        net[s] = v + b2[0];
    }
    __syncthreads();
    gemm_cbn<2>(wsW + OFF_WAKNN, wsB + BIA_AKNN, A, K3_STRIDE, K3_SLICE, 0, 128, 128, true, lane);
    __syncthreads();
    // qk, softmax over the 16 neighbor columns, weighted sum -> out
#pragma unroll
    for (int s = 0; s < 2; ++s) {
        float v = 0.f;
        const _Float16* fp = fptW + (size_t)(r0 + s) * CC + hi * 64;
        const _Float16* fk = A + s * K3_SLICE + col * K3_STRIDE + 128 + hi * 64;
        for (int i = 0; i < 64; ++i) v += (float)fk[i] * (float)fp[i];
        v += __shfl_xor(v, 16, 32);
        float qk = v * 0.08838834764831843f;   // 1/sqrt(128)
        float mx = qk;
        for (int dd = 8; dd; dd >>= 1) mx = fmaxf(mx, __shfl_xor(mx, dd, 16));
        float e = __expf(qk - mx);
        float sm = e;
        for (int dd = 8; dd; dd >>= 1) sm += __shfl_xor(sm, dd, 16);
        float r = net[s] * (e / sm);
        for (int dd = 8; dd; dd >>= 1) r += __shfl_xor(r, dd, 16);
        if (lane == 0) out[r0 + s] = r;
    }
}

// ---------------------------------------------------------------------------
extern "C" void kernel_launch(void* const* d_in, const int* in_sizes, int n_in,
                              void* d_out, int out_size, void* d_ws, size_t ws_size,
                              hipStream_t stream)
{
    const float* p       = (const float*)d_in[0];
    // d_in[1] = z (unused by the reference output path)
    const float* feat_g  = (const float*)d_in[2];
    const float* feat_pt = (const float*)d_in[3];
    const float* inputs  = (const float*)d_in[4];
    const float* blk2_w  = (const float*)d_in[40];
    const float* blk2_b  = (const float*)d_in[41];

    char* ws = (char*)d_ws;
    _Float16* wsW  = (_Float16*)ws;
    float*    wsB  = (float*)(ws + WEIGHT_BYTES);
    int*      idxK = (int*)(ws + OFFB_IDXK);
    _Float16* pdif = (_Float16*)(ws + OFFB_PDIFF);
    _Float16* vint = (_Float16*)(ws + OFFB_VINT);
    _Float16* vp   = (_Float16*)(ws + OFFB_VP);
    _Float16* fpt  = (_Float16*)(ws + OFFB_FPT);
    float* out = (float*)d_out;

    // ---- K0: fold BN + fp16 convert ----
    struct L { int base, cin, cinPad, wOff, bOff; };
    const L layers[9] = {
        {5,  3,   32,  OFF_WPD1,  BIA_PD1 },
        {10, 128, 128, OFF_WPD2,  BIA_PD2 },
        {15, 384, 384, OFF_WKNN1, BIA_KNN1},
        {20, 128, 128, OFF_WKNN2, BIA_KNN2},
        {25, 128, 128, OFF_WPT1,  BIA_PT1 },
        {30, 128, 128, OFF_WPT2,  BIA_PT2 },
        {35, 259, 288, OFF_WBLK1, BIA_BLK1},
        {42, 128, 128, OFF_WAKNN, BIA_AKNN},
        {47, 128, 128, OFF_WAPT,  BIA_APT },
    };
    for (int l = 0; l < 9; ++l) {
        int total = 128 * layers[l].cinPad;
        fold_kernel<<<(total + 255) / 256, 256, 0, stream>>>(
            (const float*)d_in[layers[l].base + 0],
            (const float*)d_in[layers[l].base + 1],
            (const float*)d_in[layers[l].base + 2],
            (const float*)d_in[layers[l].base + 3],
            (const float*)d_in[layers[l].base + 4],
            wsW + layers[l].wOff, wsB + layers[l].bOff,
            128, layers[l].cin, layers[l].cinPad);
    }

    // ---- K1: KNN + 3-NN interpolation ----
    knn_kernel<<<ROWS, 256, 0, stream>>>(p, inputs, feat_pt, idxK, pdif, vint);

    // ---- K2: per-point chain (pt1/pt2 -> vp ; attn_pt -> f_pt) ----
    pt_kernel<<<ROWS / 32, 32, 0, stream>>>(wsW, wsB, vint, vp, fpt);

    // ---- K3: fused neighbor-column chain + softmax reduce ----
    fuse_kernel<<<ROWS / 2, 32, 0, stream>>>(
        feat_g, feat_pt, blk2_w, blk2_b, wsW, wsB, idxK, pdif, vp, fpt, out);
}
